// GraphSearchPolicy_43009802502734
// MI455X (gfx1250) — matrix operand order repaired and checked by reading, not compiled, not run on hardware
//
#include <hip/hip_runtime.h>

// ---------------------------------------------------------------------------
// GraphSearchPolicy fused forward for MI455X (gfx1250, wave32, WMMA)
//
//   B=2048, A=256, DE=128, DH=256, DR=128, DX=512, ACT=256
//   out = [dist (2048*256 f32)] ++ [entropy (2048 f32)]
// ---------------------------------------------------------------------------

#define B_TOT   2048
#define A_DIM   256
#define DE      128
#define DH      256
#define DR      128
#define DX      512     // DE + DH + DR
#define ACT     256     // hidden width
#define TB      16      // batch rows per block

typedef __attribute__((ext_vector_type(8)))  float  v8f;
typedef __attribute__((ext_vector_type(16))) __bf16 v16bf;

union FragBF {            // 32-byte WMMA 16x16x32 bf16 operand fragment
    uint4 q[2];
    v16bf v;
};

__device__ __forceinline__ unsigned short f32_to_bf16(float f) {
    union { float f; unsigned int u; } x; x.f = f;
    unsigned int r = x.u + 0x7FFFu + ((x.u >> 16) & 1u);   // round-to-nearest-even
    return (unsigned short)(r >> 16);
}

// ---------------------------------------------------------------------------
// Prep: convert W1 [512,256] / W2 [256,128] (f32, row-major K x N) into
// bf16 B-fragment layout:  WB[(k16*N + col)*16 + e] = bf16(W[(k16*16+e)*N + col])
// so a WMMA lane (column = nb + (lane&15), K-half by lane>=16) loads one
// contiguous 32B chunk per k-step.
// ---------------------------------------------------------------------------
__global__ __launch_bounds__(256) void prep_weights(
    const float* __restrict__ W1, const float* __restrict__ W2,
    unsigned short* __restrict__ W1B, unsigned short* __restrict__ W2B)
{
    int id = blockIdx.x * 256 + threadIdx.x;          // grid covers 512*256
    if (id < DX * ACT) {
        int k = id >> 8;          // 0..511
        int c = id & 255;         // 0..255
        W1B[(((k >> 4) * ACT) + c) * 16 + (k & 15)] = f32_to_bf16(W1[k * ACT + c]);
    }
    if (id < ACT * DR) {
        int k = id >> 7;          // 0..255
        int c = id & 127;         // 0..127
        W2B[(((k >> 4) * DR) + c) * 16 + (k & 15)] = f32_to_bf16(W2[k * DR + c]);
    }
}

// ---------------------------------------------------------------------------
// Fused: gather/concat -> GEMM1+ReLU -> GEMM2 -> gathered logits -> masked
// softmax -> entropy.  One block = 16 batch rows, 256 threads = 8 waves.
// ---------------------------------------------------------------------------
__global__ __launch_bounds__(256) void policy_fused(
    const int*   __restrict__ e,          const int*   __restrict__ q,
    const float* __restrict__ H,          const int*   __restrict__ r_space,
    const float* __restrict__ r_mask,     const float* __restrict__ entity_emb,
    const float* __restrict__ rel_emb,
    const float* __restrict__ b1,         const float* __restrict__ b2,
    const unsigned short* __restrict__ W1B,
    const unsigned short* __restrict__ W2B,
    float* __restrict__ dist_out,         float* __restrict__ ent_out)
{
    __shared__ __align__(16) unsigned short sX0[TB * DX];   // 16 KB bf16 concat input
    __shared__ __align__(16) unsigned short sX1[TB * ACT];  //  8 KB bf16 hidden
    __shared__ __align__(16) float          sX2[TB * DR];   //  8 KB f32 policy feature
    __shared__ __align__(16) float          sLg[TB * A_DIM];// 16 KB f32 logits

    const int tid   = threadIdx.x;
    const int b0    = blockIdx.x * TB;
    const int wave  = tid >> 5;
    const int lane  = tid & 31;
    const int ln    = lane & 15;        // column within 16-wide tile
    const int khalf = lane >> 4;        // 0: K 0..15, 1: K 16..31 (A and B)

    // ---- Stage 1: X0 = concat(E[e], H, Q[q]) as bf16 in LDS -----------------
    for (int i = tid; i < TB * DX; i += 256) {
        int r = i >> 9;                 // row 0..15
        int c = i & (DX - 1);           // col 0..511
        int b = b0 + r;
        float v;
        if (c < DE)            v = entity_emb[e[b] * DE + c];
        else if (c < DE + DH)  v = H[b * DH + (c - DE)];
        else                   v = rel_emb[q[b] * DR + (c - DE - DH)];
        sX0[i] = f32_to_bf16(v);
    }
    __syncthreads();

    // ---- Stage 2: GEMM1  X1 = relu(X0 @ W1 + b1)  [16 x 256] ---------------
    {
        v8f acc0 = {};                  // n-tile = wave
        v8f acc1 = {};                  // n-tile = wave + 8
        const int n0 = wave * 16 + ln;
        const int n1 = (wave + 8) * 16 + ln;
        #pragma unroll
        for (int kt = 0; kt < DX / 32; ++kt) {
            FragBF a;
            const unsigned short* ap = &sX0[ln * DX + kt * 32 + khalf * 16];
            a.q[0] = *(const uint4*)(ap);
            a.q[1] = *(const uint4*)(ap + 8);
            const int k16 = kt * 2 + khalf;
            FragBF bf0 = *(const FragBF*)&W1B[(k16 * ACT + n0) * 16];
            FragBF bf1 = *(const FragBF*)&W1B[(k16 * ACT + n1) * 16];
            acc0 = __builtin_amdgcn_wmma_f32_16x16x32_bf16(
                       false, a.v, false, bf0.v, (short)0, acc0, false, false);
            acc1 = __builtin_amdgcn_wmma_f32_16x16x32_bf16(
                       false, a.v, false, bf1.v, (short)0, acc1, false, false);
        }
        const float bb0 = b1[n0];
        const float bb1 = b1[n1];
        #pragma unroll
        for (int r = 0; r < 8; ++r) {
            int m = r + khalf * 8;      // D layout: lanes 0-15 -> M=r, 16-31 -> M=8+r
            float v0 = acc0[r] + bb0;
            float v1 = acc1[r] + bb1;
            sX1[m * ACT + n0] = f32_to_bf16(v0 > 0.f ? v0 : 0.f);
            sX1[m * ACT + n1] = f32_to_bf16(v1 > 0.f ? v1 : 0.f);
        }
    }
    __syncthreads();

    // ---- Stage 3: GEMM2  X2 = X1 @ W2 + b2  [16 x 128] ---------------------
    {
        v8f acc = {};
        const int n = wave * 16 + ln;   // 8 waves x 16 = 128 columns
        #pragma unroll
        for (int kt = 0; kt < ACT / 32; ++kt) {
            FragBF a;
            const unsigned short* ap = &sX1[ln * ACT + kt * 32 + khalf * 16];
            a.q[0] = *(const uint4*)(ap);
            a.q[1] = *(const uint4*)(ap + 8);
            const int k16 = kt * 2 + khalf;
            FragBF bf = *(const FragBF*)&W2B[(k16 * DR + n) * 16];
            acc = __builtin_amdgcn_wmma_f32_16x16x32_bf16(
                      false, a.v, false, bf.v, (short)0, acc, false, false);
        }
        const float bb = b2[n];
        #pragma unroll
        for (int r = 0; r < 8; ++r) {
            int m = r + khalf * 8;
            sX2[m * DR + n] = acc[r] + bb;
        }
    }
    __syncthreads();

    // ---- Stage 4: gathered logits  logit[b,a] = <rel_emb[r_space[b,a]], X2[b]>
    {
        const int a = tid;              // one action column per thread
        for (int r = 0; r < TB; ++r) {
            int b   = b0 + r;
            int rel = r_space[b * A_DIM + a];
            const float4* rp = (const float4*)&rel_emb[rel * DR];
            const float4* xp = (const float4*)&sX2[r * DR];     // wave-broadcast
            float acc = 0.f;
            #pragma unroll 8
            for (int d = 0; d < DR / 4; ++d) {
                float4 rv = rp[d];
                float4 xv = xp[d];
                acc += rv.x * xv.x + rv.y * xv.y + rv.z * xv.z + rv.w * xv.w;
            }
            float mask = r_mask[b * A_DIM + a];
            sLg[r * A_DIM + a] = acc - (1.f - mask) * 1e31f;
        }
    }
    __syncthreads();

    // ---- Stage 5: softmax + entropy, one 256-wide row per wave -------------
    {
        #pragma unroll
        for (int rr = 0; rr < 2; ++rr) {
            const int r = wave + rr * 8;
            const float* lp = &sLg[r * A_DIM];
            float vals[8];
            float m = -3.402823466e38f;
            #pragma unroll
            for (int i = 0; i < 8; ++i) {
                vals[i] = lp[lane + 32 * i];
                m = fmaxf(m, vals[i]);
            }
            #pragma unroll
            for (int s = 16; s > 0; s >>= 1) m = fmaxf(m, __shfl_xor(m, s, 32));
            float sum = 0.f;
            #pragma unroll
            for (int i = 0; i < 8; ++i) { vals[i] = __expf(vals[i] - m); sum += vals[i]; }
            #pragma unroll
            for (int s = 16; s > 0; s >>= 1) sum += __shfl_xor(sum, s, 32);
            const float inv = 1.f / sum;
            float ent = 0.f;
            float* dp = &dist_out[(size_t)(b0 + r) * A_DIM];
            #pragma unroll
            for (int i = 0; i < 8; ++i) {
                float p = vals[i] * inv;
                dp[lane + 32 * i] = p;
                ent -= p * __logf(fmaxf(p, 1e-20f));
            }
            #pragma unroll
            for (int s = 16; s > 0; s >>= 1) ent += __shfl_xor(ent, s, 32);
            if (lane == 0) ent_out[b0 + r] = ent;
        }
    }
}

// ---------------------------------------------------------------------------
extern "C" void kernel_launch(void* const* d_in, const int* in_sizes, int n_in,
                              void* d_out, int out_size, void* d_ws, size_t ws_size,
                              hipStream_t stream) {
    const int*   e          = (const int*)  d_in[0];
    const int*   q          = (const int*)  d_in[1];
    const float* H          = (const float*)d_in[2];
    const int*   r_space    = (const int*)  d_in[3];
    const float* r_mask     = (const float*)d_in[4];
    const float* entity_emb = (const float*)d_in[5];
    const float* rel_emb    = (const float*)d_in[6];
    const float* W1         = (const float*)d_in[7];
    const float* b1         = (const float*)d_in[8];
    const float* W2         = (const float*)d_in[9];
    const float* b2         = (const float*)d_in[10];

    unsigned short* W1B = (unsigned short*)d_ws;            // 512*256 bf16 = 256 KB
    unsigned short* W2B = W1B + DX * ACT;                   // 256*128 bf16 =  64 KB

    float* dist_out = (float*)d_out;                        // [2048, 256]
    float* ent_out  = dist_out + (size_t)B_TOT * A_DIM;     // [2048]

    prep_weights<<<(DX * ACT + 255) / 256, 256, 0, stream>>>(W1, W2, W1B, W2B);
    policy_fused<<<B_TOT / TB, 256, 0, stream>>>(
        e, q, H, r_space, r_mask, entity_emb, rel_emb,
        b1, b2, W1B, W2B, dist_out, ent_out);
}
